// GLiNERLinkerPooler_23441931502006
// MI455X (gfx1250) — matrix operand order repaired and checked by reading, not compiled, not run on hardware
//
#include <hip/hip_runtime.h>
#include <hip/hip_bf16.h>

// ---------------------------------------------------------------------------
// GLiNER span linker/pooler for MI455X (gfx1250), wave32 + WMMA bf16.
//
// Pipeline (all on `stream`):
//  1) zero we; scatter-add hidden_states rows by words_mask (atomicAdd f32)
//  2) convert we / labels_embeds to bf16; swizzle all weight matrices into a
//     WMMA-B-fragment-native bf16 layout (one-time cost, reused by GEMMs)
//  3) tok = we@Wt+bt, lab = LE@Wl+bl           (bf16 WMMA, bf16 out)
//  4) A = t0@W1a, Bm = l0@W1b                  (bf16 WMMA, f32 out)
//  5) fused einsum kernel per (16-w-tile, c): X = t1⊙l1 -> X@W1c (WMMA),
//     + A + Bm + b1, relu, tiny @W2+b2 dot -> scores into d_out
//  6) is_start/is_end (sigmoid>=0.5 <=> score>=0); span mask/idx into d_out
//  7) P = we@Wsp1[:H], Q = we@Wsp1[H:]         (algebraic split of sp@Wsp1)
//  8) U[n] = relu(P[s_n]+Q[e_n]+bsp1) as bf16
//  9) R = U@Wsp2+bsp2 (bf16); logits = R@LE^T  -> d_out
//
// All GEMMs have K=1024, so KT=32 is a compile-time constant and every
// fragment address in the k-loop folds into instruction immediate offsets.
// Tiling is templated so every WMMA is guard-free (EXEC stays all-ones).
// B fragments for all JT n-tiles are loaded into distinct registers before
// the WMMA burst so loads pipeline with partial s_wait_loadcnt thresholds.
//
// Scratch requirement: ~160 MB in d_ws.
// ---------------------------------------------------------------------------

typedef __bf16 bf16_t;
typedef __attribute__((ext_vector_type(16))) __bf16 v16bf;
typedef __attribute__((ext_vector_type(8)))  __bf16 v8bf;
typedef __attribute__((ext_vector_type(8)))  float  v8f;

namespace {
constexpr int kT    = 3072;
constexpr int kW    = 2048;
constexpr int kH    = 1024;
constexpr int kC    = 32;
constexpr int kS    = 3;
constexpr int kHID  = 512;
constexpr int kMAXW = 12;
constexpr int kNSPAN = kW * kMAXW;          // 24576
constexpr int kKT   = kH / 32;              // 32 k-tiles (K = 1024 everywhere)
// d_out layout (floats)
constexpr size_t oScores = 0;                                   // W*C*S
constexpr size_t oIdx    = (size_t)kW * kC * kS;                // + 2*NSPAN
constexpr size_t oMask   = oIdx + 2 * (size_t)kNSPAN;           // + NSPAN
constexpr size_t oLogits = oMask + (size_t)kNSPAN;              // + NSPAN*C
}

// ---------------- fragment helpers ----------------
__device__ __forceinline__ v16bf load_frag16(const bf16_t* p0, const bf16_t* p1) {
  union { v16bf v; v8bf h[2]; } u;
  u.h[0] = *(const v8bf*)p0;
  u.h[1] = *(const v8bf*)(p1);
  return u.v;
}

// ---------------- generic bf16 WMMA GEMM (K = 1024 fixed) ----------------
// D = A(Mx1024, row-major bf16, leading dim lda) @ Bsw (swizzled 1024xN) [+bias]
// Writes f32 (Df) and/or bf16 (Db), leading dim ldd.
// WM x WN waves (WM*WN == 8), JT 16-col tiles per wave.
// Block tile: (WM*16) rows x (WN*JT*16) cols; grid must cover M,N exactly.
template <int WM, int WN, int JT>
__global__ __launch_bounds__(256)
void gemm_bf16_wmma(const bf16_t* __restrict__ A, int lda,
                    const bf16_t* __restrict__ Bsw,
                    const float*  __restrict__ bias,
                    float* __restrict__ Df, bf16_t* __restrict__ Db, int ldd)
{
  const int lane = threadIdx.x & 31;
  const int wave = threadIdx.x >> 5;
  const int row0 = blockIdx.y * (WM * 16) + (wave / WN) * 16;
  const int n0   = blockIdx.x * (WN * JT * 16) + (wave % WN) * (JT * 16);

  const int kh = (lane >> 4) << 3;             // 0 or 8
  const bf16_t* Ap = A + (size_t)(row0 + (lane & 15)) * lda + kh;
  // tile (nt,kt) lives at ((nt*kKT + kt)*32 + lane)*16
  const bf16_t* Bp = Bsw + ((size_t)(n0 >> 4) * kKT * 32 + lane) * 16;

  v8f acc[JT];
#pragma unroll
  for (int j = 0; j < JT; ++j) acc[j] = v8f{};

#pragma unroll 4
  for (int kt = 0; kt < kKT; ++kt) {
    v16bf a = load_frag16(Ap + kt * 32, Ap + kt * 32 + 16);
    v16bf b[JT];
#pragma unroll
    for (int j = 0; j < JT; ++j) {
      const bf16_t* bp = Bp + (size_t)(j * kKT + kt) * 512;
      b[j] = load_frag16(bp, bp + 8);
    }
#pragma unroll
    for (int j = 0; j < JT; ++j) {
      acc[j] = __builtin_amdgcn_wmma_f32_16x16x32_bf16(
          false, a, false, b[j], (short)0, acc[j], false, false);
    }
  }

  const int colL = lane & 15;
  const int rsub = (lane >> 4) * 8;
#pragma unroll
  for (int j = 0; j < JT; ++j) {
    int col = n0 + j * 16 + colL;
    float badd = bias ? bias[col] : 0.0f;
#pragma unroll
    for (int r = 0; r < 8; ++r) {
      int row = row0 + rsub + r;
      float v = acc[j][r] + badd;
      if (Df) Df[(size_t)row * ldd + col] = v;
      if (Db) Db[(size_t)row * ldd + col] = (bf16_t)v;
    }
  }
}

// ---------------- fused einsum + bias/relu + scores ----------------
// block = (w-tile of 16, label c). Computes h[16,512] = (t1⊙l1)@W1c + A + Bm + b1,
// relu, then scores[16,3] = h@W2 + b2 directly into d_out.
__global__ __launch_bounds__(256)
void einsum_scores_kernel(const bf16_t* __restrict__ tok_bf,   // W x 2H
                          const bf16_t* __restrict__ lab_bf,   // C x 2H
                          const bf16_t* __restrict__ W1c_sw,   // swizzled H x HID
                          const float*  __restrict__ Aw,       // W x HID
                          const float*  __restrict__ Bc,       // C x HID
                          const float*  __restrict__ b1,
                          const float*  __restrict__ W2,       // HID x S
                          const float*  __restrict__ b2,
                          float* __restrict__ scores)          // d_out
{
  __shared__ float hbuf[16][kHID];             // 32 KB
  const int w0   = blockIdx.x * 16;
  const int c    = blockIdx.y;
  const int lane = threadIdx.x & 31;
  const int wave = threadIdx.x >> 5;           // each wave owns 64 HID cols
  const int n0   = wave * 64;

  const int kh = (lane >> 4) << 3;
  const bf16_t* Ap = tok_bf + (size_t)(w0 + (lane & 15)) * (2 * kH) + kH + kh; // t1
  const bf16_t* Lp = lab_bf + (size_t)c * (2 * kH) + kH + kh;                  // l1[c]
  const bf16_t* Bp = W1c_sw + ((size_t)(n0 >> 4) * kKT * 32 + lane) * 16;

  v8f acc[4];
#pragma unroll
  for (int j = 0; j < 4; ++j) acc[j] = v8f{};

#pragma unroll 4
  for (int kt = 0; kt < kKT; ++kt) {
    v16bf t = load_frag16(Ap + kt * 32, Ap + kt * 32 + 16);
    v16bf l = load_frag16(Lp + kt * 32, Lp + kt * 32 + 16);
    v16bf b[4];
#pragma unroll
    for (int j = 0; j < 4; ++j) {
      const bf16_t* bp = Bp + (size_t)(j * kKT + kt) * 512;
      b[j] = load_frag16(bp, bp + 8);
    }
    v16bf x;
#pragma unroll
    for (int q = 0; q < 16; ++q) x[q] = (bf16_t)((float)t[q] * (float)l[q]);
#pragma unroll
    for (int j = 0; j < 4; ++j) {
      acc[j] = __builtin_amdgcn_wmma_f32_16x16x32_bf16(
          false, x, false, b[j], (short)0, acc[j], false, false);
    }
  }

  const int colL = lane & 15;
  const int rsub = (lane >> 4) * 8;
#pragma unroll
  for (int j = 0; j < 4; ++j) {
    int n = n0 + j * 16 + colL;
#pragma unroll
    for (int r = 0; r < 8; ++r) {
      int ri = rsub + r;
      float v = acc[j][r] + Aw[(size_t)(w0 + ri) * kHID + n]
                          + Bc[(size_t)c * kHID + n] + b1[n];
      hbuf[ri][n] = fmaxf(v, 0.0f);
    }
  }
  __syncthreads();
  for (int idx = threadIdx.x; idx < 16 * kS; idx += 256) {
    int i = idx / kS, s = idx - i * kS;
    float sum = b2[s];
    for (int d = 0; d < kHID; ++d) sum += hbuf[i][d] * W2[d * kS + s];
    scores[((size_t)(w0 + i) * kC + c) * kS + s] = sum;
  }
}

// ---------------- elementwise / prep kernels ----------------
__global__ void zero_f32_kernel(float* __restrict__ p, size_t n) {
  size_t i = (size_t)blockIdx.x * blockDim.x + threadIdx.x;
  if (i < n) p[i] = 0.0f;
}

__global__ void scatter_tokens_kernel(const float* __restrict__ hs,
                                      const int* __restrict__ mask,
                                      float* __restrict__ we) {
  size_t i = (size_t)blockIdx.x * blockDim.x + threadIdx.x;
  if (i >= (size_t)kT * kH) return;
  int t = (int)(i / kH);
  int h = (int)(i - (size_t)t * kH);
  int m = mask[t];
  if (m > 0) atomicAdd(&we[(size_t)(m - 1) * kH + h], hs[i]);
}

__global__ void f32_to_bf16_kernel(const float* __restrict__ x,
                                   bf16_t* __restrict__ y, size_t n) {
  size_t i = (size_t)blockIdx.x * blockDim.x + threadIdx.x;
  if (i < n) y[i] = (bf16_t)x[i];
}

// Swizzle f32 B (KxN row-major) into WMMA-B-fragment-native bf16 layout:
// Bsw[((nt*KT + kt)*32 + lane)*16 + e] = B[kt*32 + (lane>>4)*8 + (e&7) + (e>>3)*16, nt*16 + (lane&15)]
__global__ void swizzle_b_kernel(const float* __restrict__ B,
                                 bf16_t* __restrict__ Bsw, int K, int N) {
  size_t i = (size_t)blockIdx.x * blockDim.x + threadIdx.x;
  if (i >= (size_t)K * N) return;
  int e    = (int)(i & 15);
  int lane = (int)((i >> 4) & 31);
  size_t t = i >> 9;
  int KT = K >> 5;
  int kt = (int)(t % KT);
  int nt = (int)(t / KT);
  int col = nt * 16 + (lane & 15);
  int k   = kt * 32 + (lane >> 4) * 8 + (e & 7) + (e >> 3) * 16;
  Bsw[i] = (bf16_t)B[(size_t)k * N + col];
}

// Same, but the source is B^T stored as N x K row-major (used for LE^T).
__global__ void swizzle_bt_kernel(const float* __restrict__ Bt,
                                  bf16_t* __restrict__ Bsw, int K, int N) {
  size_t i = (size_t)blockIdx.x * blockDim.x + threadIdx.x;
  if (i >= (size_t)K * N) return;
  int e    = (int)(i & 15);
  int lane = (int)((i >> 4) & 31);
  size_t t = i >> 9;
  int KT = K >> 5;
  int kt = (int)(t % KT);
  int nt = (int)(t / KT);
  int col = nt * 16 + (lane & 15);
  int k   = kt * 32 + (lane >> 4) * 8 + (e & 7) + (e >> 3) * 16;
  Bsw[i] = (bf16_t)Bt[(size_t)col * K + k];
}

__global__ void start_end_kernel(const float* __restrict__ scores,
                                 int* __restrict__ is_start,
                                 int* __restrict__ is_end) {
  int w = blockIdx.x * blockDim.x + threadIdx.x;
  if (w >= kW) return;
  int st = 0, en = 0;
  for (int c = 0; c < kC; ++c) {
    const float* p = scores + ((size_t)w * kC + c) * kS;
    st |= (p[0] >= 0.0f);                     // sigmoid(x)>=0.5 <=> x>=0
    en |= (p[1] >= 0.0f);
  }
  is_start[w] = st;
  is_end[w] = en;
}

__global__ void spans_kernel(const int* __restrict__ is_start,
                             const int* __restrict__ is_end,
                             int* __restrict__ s_idx, int* __restrict__ e_idx,
                             float* __restrict__ out_idx,    // d_out + oIdx
                             float* __restrict__ out_mask) { // d_out + oMask
  int n = blockIdx.x * blockDim.x + threadIdx.x;
  if (n >= kNSPAN) return;
  int st  = n / kMAXW;
  int off = n - st * kMAXW;
  int er  = st + off;
  int valid = (er < kW);
  int e = er < kW ? er : (kW - 1);
  int m = valid && is_start[st] && is_end[e];
  s_idx[n] = m ? st : 0;
  e_idx[n] = m ? e : 0;
  out_idx[2 * n + 0] = (float)st;
  out_idx[2 * n + 1] = (float)e;
  out_mask[n] = m ? 1.0f : 0.0f;
}

__global__ void build_u_kernel(const float* __restrict__ P,
                               const float* __restrict__ Q,
                               const float* __restrict__ bsp1,
                               const int* __restrict__ s_idx,
                               const int* __restrict__ e_idx,
                               bf16_t* __restrict__ U) {
  size_t i = (size_t)blockIdx.x * blockDim.x + threadIdx.x;
  if (i >= (size_t)kNSPAN * kH) return;
  int n = (int)(i / kH);
  int d = (int)(i - (size_t)n * kH);
  float v = P[(size_t)s_idx[n] * kH + d] + Q[(size_t)e_idx[n] * kH + d] + bsp1[d];
  U[i] = (bf16_t)fmaxf(v, 0.0f);
}

// ---------------------------------------------------------------------------
extern "C" void kernel_launch(void* const* d_in, const int* in_sizes, int n_in,
                              void* d_out, int out_size, void* d_ws, size_t ws_size,
                              hipStream_t stream) {
  const float* hs   = (const float*)d_in[0];
  const int*   wm   = (const int*)d_in[1];
  const float* LE   = (const float*)d_in[2];
  const float* Wt   = (const float*)d_in[3];
  const float* bt   = (const float*)d_in[4];
  const float* Wl   = (const float*)d_in[5];
  const float* bl   = (const float*)d_in[6];
  const float* W1a  = (const float*)d_in[7];
  const float* W1b  = (const float*)d_in[8];
  const float* W1c  = (const float*)d_in[9];
  const float* b1   = (const float*)d_in[10];
  const float* W2   = (const float*)d_in[11];
  const float* b2   = (const float*)d_in[12];
  const float* Wsp1 = (const float*)d_in[13];
  const float* bsp1 = (const float*)d_in[14];
  const float* Wsp2 = (const float*)d_in[15];
  const float* bsp2 = (const float*)d_in[16];
  float* out = (float*)d_out;

  // ---- carve workspace (deterministic) ----
  char* wsp = (char*)d_ws;
  auto carve = [&](size_t bytes) -> void* {
    void* p = (void*)wsp;
    wsp += (bytes + 255) & ~(size_t)255;
    return p;
  };
  float*  we      = (float*)carve((size_t)kW * kH * 4);
  bf16_t* we_bf   = (bf16_t*)carve((size_t)kW * kH * 2);
  bf16_t* tok_bf  = (bf16_t*)carve((size_t)kW * 2 * kH * 2);
  bf16_t* lab_bf  = (bf16_t*)carve((size_t)kC * 2 * kH * 2);
  bf16_t* le_bf   = (bf16_t*)carve((size_t)kC * kH * 2);
  float*  Amat    = (float*)carve((size_t)kW * kHID * 4);
  float*  Bmat    = (float*)carve((size_t)kC * kHID * 4);
  bf16_t* Wt_sw   = (bf16_t*)carve((size_t)kH * 2 * kH * 2);
  bf16_t* Wl_sw   = (bf16_t*)carve((size_t)kH * 2 * kH * 2);
  bf16_t* W1a_sw  = (bf16_t*)carve((size_t)kH * kHID * 2);
  bf16_t* W1b_sw  = (bf16_t*)carve((size_t)kH * kHID * 2);
  bf16_t* W1c_sw  = (bf16_t*)carve((size_t)kH * kHID * 2);
  bf16_t* Wsp1t_sw= (bf16_t*)carve((size_t)kH * kH * 2);
  bf16_t* Wsp1b_sw= (bf16_t*)carve((size_t)kH * kH * 2);
  bf16_t* Wsp2_sw = (bf16_t*)carve((size_t)kH * kH * 2);
  bf16_t* LEt_sw  = (bf16_t*)carve((size_t)kH * kC * 2);
  int*    is_start= (int*)carve((size_t)kW * 4);
  int*    is_end  = (int*)carve((size_t)kW * 4);
  int*    s_idx   = (int*)carve((size_t)kNSPAN * 4);
  int*    e_idx   = (int*)carve((size_t)kNSPAN * 4);
  float*  Pm      = (float*)carve((size_t)kW * kH * 4);
  float*  Qm      = (float*)carve((size_t)kW * kH * 4);
  bf16_t* Um      = (bf16_t*)carve((size_t)kNSPAN * kH * 2);
  bf16_t* Rm      = (bf16_t*)carve((size_t)kNSPAN * kH * 2);
  (void)ws_size; (void)in_sizes; (void)n_in; (void)out_size;

  auto blocks = [](size_t n) { return (unsigned)((n + 255) / 256); };

  // 1) word embeddings via scatter-add
  zero_f32_kernel<<<blocks((size_t)kW * kH), 256, 0, stream>>>(we, (size_t)kW * kH);
  scatter_tokens_kernel<<<blocks((size_t)kT * kH), 256, 0, stream>>>(hs, wm, we);
  f32_to_bf16_kernel<<<blocks((size_t)kW * kH), 256, 0, stream>>>(we, we_bf, (size_t)kW * kH);
  f32_to_bf16_kernel<<<blocks((size_t)kC * kH), 256, 0, stream>>>(LE, le_bf, (size_t)kC * kH);

  // 2) weight swizzles (bf16, WMMA-B-native)
  swizzle_b_kernel<<<blocks((size_t)kH * 2 * kH), 256, 0, stream>>>(Wt, Wt_sw, kH, 2 * kH);
  swizzle_b_kernel<<<blocks((size_t)kH * 2 * kH), 256, 0, stream>>>(Wl, Wl_sw, kH, 2 * kH);
  swizzle_b_kernel<<<blocks((size_t)kH * kHID), 256, 0, stream>>>(W1a, W1a_sw, kH, kHID);
  swizzle_b_kernel<<<blocks((size_t)kH * kHID), 256, 0, stream>>>(W1b, W1b_sw, kH, kHID);
  swizzle_b_kernel<<<blocks((size_t)kH * kHID), 256, 0, stream>>>(W1c, W1c_sw, kH, kHID);
  swizzle_b_kernel<<<blocks((size_t)kH * kH), 256, 0, stream>>>(Wsp1, Wsp1t_sw, kH, kH);
  swizzle_b_kernel<<<blocks((size_t)kH * kH), 256, 0, stream>>>(Wsp1 + (size_t)kH * kH, Wsp1b_sw, kH, kH);
  swizzle_b_kernel<<<blocks((size_t)kH * kH), 256, 0, stream>>>(Wsp2, Wsp2_sw, kH, kH);
  swizzle_bt_kernel<<<blocks((size_t)kH * kC), 256, 0, stream>>>(LE, LEt_sw, kH, kC);

  // Standard GEMM: 32x256 block tile (2x4 waves, 4 tiles/wave). N%256==0, M%32==0.
  auto gemm = [&](const bf16_t* A, int lda, const bf16_t* Bsw, const float* bias,
                  float* Df, bf16_t* Db, int ldd, int M, int N) {
    dim3 g((unsigned)(N / 256), (unsigned)(M / 32));
    gemm_bf16_wmma<2, 4, 4><<<g, 256, 0, stream>>>(A, lda, Bsw, bias, Df, Db, ldd);
  };

  // 3) tok / lab projections (bf16 out; t0 = cols [0,H), t1 = cols [H,2H))
  gemm(we_bf, kH, Wt_sw, bt, nullptr, tok_bf, 2 * kH, kW, 2 * kH);
  gemm(le_bf, kH, Wl_sw, bl, nullptr, lab_bf, 2 * kH, kC, 2 * kH);

  // 4) A = t0@W1a, Bm = l0@W1b (f32 out, bias folded later with b1)
  gemm(tok_bf, 2 * kH, W1a_sw, nullptr, Amat, nullptr, kHID, kW, kHID);
  gemm(lab_bf, 2 * kH, W1b_sw, nullptr, Bmat, nullptr, kHID, kC, kHID);

  // 5) fused einsum + relu + scores
  einsum_scores_kernel<<<dim3(kW / 16, kC), 256, 0, stream>>>(
      tok_bf, lab_bf, W1c_sw, Amat, Bmat, b1, W2, b2, out + oScores);

  // 6) start/end flags and span mask/idx
  start_end_kernel<<<blocks(kW), 256, 0, stream>>>(out + oScores, is_start, is_end);
  spans_kernel<<<blocks(kNSPAN), 256, 0, stream>>>(is_start, is_end, s_idx, e_idx,
                                                   out + oIdx, out + oMask);

  // 7) P/Q precompute (algebraic split of sp@Wsp1)
  gemm(we_bf, kH, Wsp1t_sw, nullptr, Pm, nullptr, kH, kW, kH);
  gemm(we_bf, kH, Wsp1b_sw, nullptr, Qm, nullptr, kH, kW, kH);

  // 8) U = relu(P[s]+Q[e]+bsp1) as bf16
  build_u_kernel<<<blocks((size_t)kNSPAN * kH), 256, 0, stream>>>(Pm, Qm, bsp1, s_idx, e_idx, Um);

  // 9) R = U@Wsp2 + bsp2 (bf16)
  gemm(Um, kH, Wsp2_sw, bsp2, nullptr, Rm, kH, kNSPAN, kH);

  // 10) logits = R@LE^T -> d_out.  N=32: 128x32 block tile (8x1 waves, 2 tiles/wave).
  {
    dim3 g(1u, (unsigned)(kNSPAN / 128));
    gemm_bf16_wmma<8, 1, 2><<<g, 256, 0, stream>>>(Rm, kH, LEt_sw, nullptr,
                                                   out + oLogits, nullptr, kC);
  }
}
